// EdgeConv_55482387529806
// MI455X (gfx1250) — matrix operand order repaired and checked by reading, not compiled, not run on hardware
//
#include <hip/hip_runtime.h>
#include <hip/hip_bf16.h>

typedef __attribute__((ext_vector_type(16))) _Float16 v16h;
typedef __attribute__((ext_vector_type(8)))  float    v8f;

#define IN_CH   32
#define OUT_CH  64
#define KDIM    64          // 2*IN_CH
#define TILE_M  16
#define WAVES_PER_BLOCK 8
#define BLOCK_THREADS   256
#define FEAT_LD 72          // padded LDS row stride (halves) for edge-feature tiles
#define W_LD    68          // padded LDS row stride (halves) for W
#define GEMM_BLOCKS 512

union FragH { v16h v; unsigned u[8]; };
union FragF { v8f  v; float   f[8]; };

__device__ __forceinline__ unsigned packrtz(float a, float b) {
    // one v_cvt_pk_rtz_f16_f32
    return __builtin_bit_cast(unsigned, __builtin_amdgcn_cvt_pkrtz(a, b));
}

// ---------------------------------------------------------------------------
// init: zero the max-keys (in d_out) and the BN accumulators (ws[0..127])
// ---------------------------------------------------------------------------
__global__ void ec_init_kernel(unsigned* __restrict__ outkeys, int nOut,
                               float* __restrict__ stats) {
    int stride = gridDim.x * blockDim.x;
    for (int i = blockIdx.x * blockDim.x + threadIdx.x; i < nOut; i += stride)
        outkeys[i] = 0u;
    int t = blockIdx.x * blockDim.x + threadIdx.x;
    if (t < 128) stats[t] = 0.0f;
}

// ---------------------------------------------------------------------------
// fused gather + WMMA GEMM. PASS==0: accumulate BN sum/sumsq.
// PASS==1: apply BN affine and scatter-max (monotonic-uint atomicMax).
// ---------------------------------------------------------------------------
template <int PASS>
__global__ __launch_bounds__(BLOCK_THREADS)
void ec_gemm_kernel(const float* __restrict__ x,
                    const int* __restrict__ ei32,       // low dwords of int64 [2,E]
                    const float* __restrict__ W,        // [64, 64] row-major
                    const float* __restrict__ bias,     // [64]
                    float* __restrict__ stat_sum,       // [64]
                    float* __restrict__ stat_sq,        // [64]
                    const float* __restrict__ scale,    // [64] (pass 1)
                    const float* __restrict__ shift,    // [64] (pass 1)
                    unsigned* __restrict__ outkeys,     // [N, 64] as keys
                    int E) {
    __shared__ _Float16 Wsh[OUT_CH * W_LD];
    __shared__ _Float16 feat[WAVES_PER_BLOCK][TILE_M * FEAT_LD];
    __shared__ int      srcIdx[WAVES_PER_BLOCK][TILE_M];
    __shared__ float    biasSh[OUT_CH];
    __shared__ float    scaleSh[OUT_CH];
    __shared__ float    shiftSh[OUT_CH];

    const int tid = threadIdx.x;
    // stage W (as f16; layout Wsh[n][k] == W[n][k] -> B[k][n] fragment source)
    for (int i = tid; i < OUT_CH * KDIM; i += BLOCK_THREADS) {
        int o = i >> 6, f = i & 63;
        Wsh[o * W_LD + f] = (_Float16)W[i];
    }
    if (tid < OUT_CH) {
        biasSh[tid] = bias[tid];
        if (PASS == 1) { scaleSh[tid] = scale[tid]; shiftSh[tid] = shift[tid]; }
    }
    __syncthreads();

    const int lane  = tid & 31;
    const int g     = (lane >= 16) ? 1 : 0;   // half-wave group
    const int nbase = lane & 15;
    const int m     = lane & 15;              // edge row inside tile
    const int mofs  = g * 8;

    // B fragments (wave-invariant): element j -> K = kh*32 + g*16 + j,
    // N = nt*16 + nbase; dword pairs are consecutive K -> b128-coalescible.
    FragH Bfrag[4][2];
    for (int nt = 0; nt < 4; ++nt) {
        int n = nt * 16 + nbase;
        for (int kh = 0; kh < 2; ++kh) {
            const unsigned* src = (const unsigned*)&Wsh[n * W_LD + kh * 32 + g * 16];
            #pragma unroll
            for (int i = 0; i < 8; ++i) Bfrag[nt][kh].u[i] = src[i];
        }
    }

    // hoist per-lane channel constants out of the tile loop (regs, not LDS)
    float bReg[4], scReg[4], shReg[4];
    #pragma unroll
    for (int nt = 0; nt < 4; ++nt) {
        int ch = nt * 16 + nbase;
        bReg[nt] = biasSh[ch];
        if (PASS == 1) { scReg[nt] = scaleSh[ch]; shReg[nt] = shiftSh[ch]; }
    }

    const int gw     = blockIdx.x * WAVES_PER_BLOCK + (tid >> 5);
    const int nWaves = gridDim.x * WAVES_PER_BLOCK;
    const int tiles  = (E + TILE_M - 1) / TILE_M;

    _Float16* myFeat = feat[tid >> 5];
    int*      mySrc  = srcIdx[tid >> 5];

    float sumAcc[4] = {0.f, 0.f, 0.f, 0.f};
    float sqAcc[4]  = {0.f, 0.f, 0.f, 0.f};

    for (int t = gw; t < tiles; t += nWaves) {
        // e0 is uniform across the wave; force it scalar so the epilogue
        // `full` check is a pure s_cmp/s_cbranch (no exec save/restore).
        const int e0 = __builtin_amdgcn_readfirstlane(t * TILE_M);
        int e  = e0 + m;
        int ec = (e < E) ? e : (E - 1);

        // prefetch next tile's indices (speculative; safe if OOB)
        __builtin_prefetch(&ei32[2 * (ec + nWaves * TILE_M)], 0, 1);

        int s  = ei32[2 * ec];            // low dword of int64 source
        int tg = ei32[2 * (E + ec)];      // low dword of int64 target
        if (g == 0) mySrc[m] = s;

        // each lane gathers its 16-channel half of x_i / x_j (128-bit loads)
        const float4* xi4 = (const float4*)(x + (long long)s  * IN_CH + g * 16);
        const float4* xj4 = (const float4*)(x + (long long)tg * IN_CH + g * 16);
        float4 vi[4], vj[4];
        #pragma unroll
        for (int q = 0; q < 4; ++q) { vi[q] = xi4[q]; vj[q] = xj4[q]; }

        // feat row m: [0..31] = x_i, [32..63] = x_j - x_i  (f16, packed dwords)
        unsigned* rowXi = (unsigned*)&myFeat[m * FEAT_LD + g * 16];
        unsigned* rowDj = (unsigned*)&myFeat[m * FEAT_LD + 32 + g * 16];
        #pragma unroll
        for (int q = 0; q < 4; ++q) {
            rowXi[q * 2 + 0] = packrtz(vi[q].x, vi[q].y);
            rowXi[q * 2 + 1] = packrtz(vi[q].z, vi[q].w);
            rowDj[q * 2 + 0] = packrtz(vj[q].x - vi[q].x, vj[q].y - vi[q].y);
            rowDj[q * 2 + 1] = packrtz(vj[q].z - vi[q].z, vj[q].w - vi[q].w);
        }

        // A fragments (16x32, contiguous K order chosen here, not the ISA table order):
        // element j: K = kh*32 + g*16 + j (matches B fragment K order)
        // -> one contiguous 16-half run per kh (compiler fuses to ds_load_b128)
        FragH Afrag[2];
        #pragma unroll
        for (int kh = 0; kh < 2; ++kh) {
            #pragma unroll
            for (int i = 0; i < 8; ++i) {
                int kk = kh * 32 + g * 16 + 2 * i;
                Afrag[kh].u[i] = *(const unsigned*)&myFeat[m * FEAT_LD + kk];
            }
        }

        // 8 WMMAs: h[16x64] = feat[16x64] @ W^T  (K split into two 32-chunks)
        FragF acc[4];
        #pragma unroll
        for (int nt = 0; nt < 4; ++nt) {
            v8f c = {};
            c = __builtin_amdgcn_wmma_f32_16x16x32_f16(
                    false, Afrag[0].v, false, Bfrag[nt][0].v, (short)0, c, false, false);
            c = __builtin_amdgcn_wmma_f32_16x16x32_f16(
                    false, Afrag[1].v, false, Bfrag[nt][1].v, (short)0, c, false, false);
            acc[nt].v = c;
        }

        // epilogue. C/D layout: VGPR r -> M = r + g*8, N = nt*16 + nbase.
        const bool full = (e0 + TILE_M) <= E;   // scalar -> s_cbranch
        if (PASS == 0) {
            if (full) {
                #pragma unroll
                for (int nt = 0; nt < 4; ++nt) {
                    #pragma unroll
                    for (int r = 0; r < 8; ++r) {
                        float v = fmaxf(acc[nt].f[r] + bReg[nt], 0.f);
                        sumAcc[nt] += v;
                        sqAcc[nt]  += v * v;
                    }
                }
            } else {
                #pragma unroll
                for (int nt = 0; nt < 4; ++nt) {
                    #pragma unroll
                    for (int r = 0; r < 8; ++r) {
                        if (e0 + r + mofs < E) {
                            float v = fmaxf(acc[nt].f[r] + bReg[nt], 0.f);
                            sumAcc[nt] += v;
                            sqAcc[nt]  += v * v;
                        }
                    }
                }
            }
        } else {
            if (full) {
                #pragma unroll
                for (int r = 0; r < 8; ++r) {
                    unsigned base = (unsigned)mySrc[r + mofs] * OUT_CH + nbase;
                    #pragma unroll
                    for (int nt = 0; nt < 4; ++nt) {
                        float v = fmaxf(acc[nt].f[r] + bReg[nt], 0.f);
                        float y = v * scReg[nt] + shReg[nt];
                        unsigned bits = __float_as_uint(y);
                        unsigned key  = (bits & 0x80000000u) ? ~bits : (bits | 0x80000000u);
                        atomicMax(&outkeys[base + nt * 16], key);
                    }
                }
            } else {
                #pragma unroll
                for (int r = 0; r < 8; ++r) {
                    if (e0 + r + mofs < E) {
                        unsigned base = (unsigned)mySrc[r + mofs] * OUT_CH + nbase;
                        #pragma unroll
                        for (int nt = 0; nt < 4; ++nt) {
                            float v = fmaxf(acc[nt].f[r] + bReg[nt], 0.f);
                            float y = v * scReg[nt] + shReg[nt];
                            unsigned bits = __float_as_uint(y);
                            unsigned key  = (bits & 0x80000000u) ? ~bits : (bits | 0x80000000u);
                            atomicMax(&outkeys[base + nt * 16], key);
                        }
                    }
                }
            }
        }
    }

    if (PASS == 0) {
        #pragma unroll
        for (int nt = 0; nt < 4; ++nt) {
            int ch = nt * 16 + nbase;
            atomicAdd(&stat_sum[ch], sumAcc[nt]);
            atomicAdd(&stat_sq[ch],  sqAcc[nt]);
        }
    }
}

// ---------------------------------------------------------------------------
// BN stats -> affine (scale, shift)
// ---------------------------------------------------------------------------
__global__ void ec_finalize_stats(const float* __restrict__ sum,
                                  const float* __restrict__ sq,
                                  const float* __restrict__ gamma,
                                  const float* __restrict__ beta,
                                  float* __restrict__ scale,
                                  float* __restrict__ shift,
                                  float invE) {
    int c = threadIdx.x;
    if (c < OUT_CH) {
        float mu  = sum[c] * invE;
        float var = sq[c] * invE - mu * mu;
        float is  = rsqrtf(var + 1e-5f);
        float sc  = gamma[c] * is;
        scale[c]  = sc;
        shift[c]  = beta[c] - mu * sc;
    }
}

// ---------------------------------------------------------------------------
// decode monotonic keys back to floats; key==0 means empty segment -> 0.0
// ---------------------------------------------------------------------------
__global__ void ec_finalize_out(unsigned* __restrict__ outkeys, int nOut) {
    int stride = gridDim.x * blockDim.x;
    for (int i = blockIdx.x * blockDim.x + threadIdx.x; i < nOut; i += stride) {
        unsigned k = outkeys[i];
        float y;
        if (k == 0u)              y = 0.0f;
        else if (k & 0x80000000u) y = __uint_as_float(k & 0x7FFFFFFFu);
        else                      y = __uint_as_float(~k);
        ((float*)outkeys)[i] = y;
    }
}

// ---------------------------------------------------------------------------
extern "C" void kernel_launch(void* const* d_in, const int* in_sizes, int n_in,
                              void* d_out, int out_size, void* d_ws, size_t ws_size,
                              hipStream_t stream) {
    const float* x     = (const float*)d_in[0];
    const int*   ei32  = (const int*)d_in[1];   // int64 viewed as dword pairs
    const float* W     = (const float*)d_in[2];
    const float* b     = (const float*)d_in[3];
    const float* gamma = (const float*)d_in[4];
    const float* beta  = (const float*)d_in[5];

    int E    = in_sizes[1] / 2;
    int nOut = out_size;

    float*    ws      = (float*)d_ws;
    float*    w_sum   = ws;          // [64]
    float*    w_sq    = ws + 64;     // [64]
    float*    w_scale = ws + 128;    // [64]
    float*    w_shift = ws + 192;    // [64]
    unsigned* outk    = (unsigned*)d_out;

    ec_init_kernel<<<2048, 256, 0, stream>>>(outk, nOut, ws);

    ec_gemm_kernel<0><<<GEMM_BLOCKS, BLOCK_THREADS, 0, stream>>>(
        x, ei32, W, b, w_sum, w_sq, w_scale, w_shift, outk, E);

    ec_finalize_stats<<<1, 64, 0, stream>>>(w_sum, w_sq, gamma, beta,
                                            w_scale, w_shift, 1.0f / (float)E);

    ec_gemm_kernel<1><<<GEMM_BLOCKS, BLOCK_THREADS, 0, stream>>>(
        x, ei32, W, b, w_sum, w_sq, w_scale, w_shift, outk, E);

    ec_finalize_out<<<2048, 256, 0, stream>>>(outk, nOut);
}
